// PlanningDecoder_origin_34754875359809
// MI455X (gfx1250) — matrix operand order repaired and checked by reading, not compiled, not run on hardware
//
#include <hip/hip_runtime.h>
#include <hip/hip_bf16.h>

// ---------------------------------------------------------------------------
// PlanningDecoder for MI455X (gfx1250, wave32, WMMA bf16 16x16x32).
//
// Input pointer order assumption (depth-first insertion order of the
// setup_inputs() dict):
//   0 tgt (32,128,6,256) f32   1 memory (32,512,256) f32   2 m_pos (1,6,256) f32
//   3..10  r2r  : Wq Wk Wv Wo bq bk bv bo
//   11..18 m2m  : Wq Wk Wv Wo bq bk bv bo
//   19..26 cross: Wq Wk Wv Wo bq bk bv bo
//   27 W1 28 b1 29 W2 30 b2
//   31/32 ln1 g,b  33/34 ln2  35/36 ln3  37/38 ln4
//   39 tgt_key_padding_mask (32,128) u8   40 memory_key_padding_mask (32,512) u8
// ---------------------------------------------------------------------------

#define BSZ 32
#define RR  128
#define MMD 6
#define DD  256
#define HHD 8
#define NNM 512
#define DH  32
#define TT  (BSZ*RR*MMD)   // 24576 decoder tokens
#define TMEM (BSZ*NNM)     // 16384 memory tokens
#define MLPD 1024

typedef __bf16 bf16;
typedef __attribute__((ext_vector_type(16))) __bf16 bf16x16;
typedef __attribute__((ext_vector_type(8)))  float  floatx8;

union Frag {
    bf16x16 v;
    float4  f4[2];
};

// ---------------------------------------------------------------------------
// Weight prep: fp32 [K][N] -> bf16 [N][K] (B operand wants N-major rows so a
// B fragment is two contiguous 16B loads per lane).
// ---------------------------------------------------------------------------
__global__ void wconv_T(const float* __restrict__ W, bf16* __restrict__ Bt,
                        int K, int Nn) {
    int idx = blockIdx.x * 256 + threadIdx.x;
    if (idx >= K * Nn) return;
    int k = idx / Nn, n = idx - k * Nn;
    Bt[(size_t)n * K + k] = (bf16)W[idx];
}

__global__ void f2bf(const float* __restrict__ src, bf16* __restrict__ dst, int n) {
    int i = blockIdx.x * 256 + threadIdx.x;
    if (i < n) dst[i] = (bf16)src[i];
}

// ---------------------------------------------------------------------------
// LayerNorm: one wave per token (D=256 -> 8 values/lane). Optional stage-1
// gather (token reorder (b,r,m)->(b,m,r)) and optional second output with
// m_pos added (for the m2m q/k path).
// ---------------------------------------------------------------------------
__global__ void ln_fused(const float* __restrict__ X, const float* __restrict__ g,
                         const float* __restrict__ bb, bf16* __restrict__ out,
                         bf16* __restrict__ out2, const float* __restrict__ pos,
                         int gatherS1) {
    const int wave = threadIdx.x >> 5, lane = threadIdx.x & 31;
    const int s = blockIdx.x * 8 + wave;
    int c = s;
    if (gatherS1) {
        int b = s / (MMD * RR);
        int rem = s - b * (MMD * RR);
        int mi = rem / RR;
        int r = rem - mi * RR;
        c = (b * RR + r) * MMD + mi;
    }
    const float* xr = X + (size_t)c * DD;
    float xv[8], sum = 0.f;
#pragma unroll
    for (int j = 0; j < 8; ++j) { xv[j] = xr[j * 32 + lane]; sum += xv[j]; }
#pragma unroll
    for (int m = 16; m >= 1; m >>= 1) sum += __shfl_xor(sum, m);
    const float mean = sum * (1.f / DD);
    float var = 0.f;
#pragma unroll
    for (int j = 0; j < 8; ++j) { float d = xv[j] - mean; var += d * d; }
#pragma unroll
    for (int m = 16; m >= 1; m >>= 1) var += __shfl_xor(var, m);
    const float inv = rsqrtf(var * (1.f / DD) + 1e-5f);
    bf16* orow  = out + (size_t)s * DD;
    bf16* o2row = out2 ? out2 + (size_t)s * DD : nullptr;
    const float* prow = pos ? pos + (size_t)(s % MMD) * DD : nullptr;
#pragma unroll
    for (int j = 0; j < 8; ++j) {
        int d = j * 32 + lane;
        float y = (xv[j] - mean) * inv * g[d] + bb[d];
        orow[d] = (bf16)y;
        if (o2row) o2row[d] = (bf16)(y + prow[d]);
    }
}

// ---------------------------------------------------------------------------
// WMMA GEMM: Out[M,N] = A[M,K](bf16) * Bt[N,K]^T(bf16) + bias, optional ReLU,
// fp32 or bf16 output.
// Block = 8 waves in a 2x4 grid -> 32-row x 256-col block tile.
// Each wave computes 16x64 (4 WMMA tiles) so one A fragment feeds 4 WMMAs.
// A is staged in LDS in 32x256 K-slabs (16KB) so LDS use is K-independent.
// ---------------------------------------------------------------------------
#define KC 256
__global__ void gemm_bf16_wmma(const bf16* __restrict__ A, const bf16* __restrict__ Bt,
                               const float* __restrict__ bias,
                               float* __restrict__ outF, bf16* __restrict__ outH,
                               int K, int Nn, int relu) {
    __shared__ __align__(16) bf16 smA[32 * KC];
    const int tid = threadIdx.x;
    const size_t aRow0 = (size_t)blockIdx.x * 32;
    const int wave = tid >> 5, lane = tid & 31, lh = lane & 15, half = lane >> 4;
    const int rw = wave >> 2;                   // row-wave: 0..1
    const int cw = wave & 3;                    // col-wave: 0..3
    const int nBase = blockIdx.y * 256 + cw * 64;

    floatx8 acc[4];
#pragma unroll
    for (int j = 0; j < 4; ++j) acc[j] = (floatx8){0.f,0.f,0.f,0.f,0.f,0.f,0.f,0.f};

    for (int k0 = 0; k0 < K; k0 += KC) {
        __syncthreads();
        // stage A slab: 32 rows x 256 k (16KB), 4 float4 chunks per thread
        for (int i = tid; i < 32 * (KC >> 3); i += 256) {
            int row = i >> 5, seg = i & 31;     // KC/8 = 32 chunks per row
            ((float4*)smA)[i] = *(const float4*)(A + (aRow0 + row) * K + k0 + seg * 8);
        }
        __syncthreads();
#pragma unroll 4
        for (int kt = 0; kt < (KC >> 5); ++kt) {
            Frag a;
            const int abase = (rw * 16 + lh) * KC + kt * 32 + half * 16;
            a.f4[0] = ((const float4*)smA)[abase >> 3];
            a.f4[1] = ((const float4*)smA)[(abase >> 3) + 1];
#pragma unroll
            for (int j = 0; j < 4; ++j) {
                Frag b;
                const bf16* bRow = Bt + (size_t)(nBase + j * 16 + lh) * K + k0 + kt * 32;
                const float4* bp = (const float4*)(bRow + half * 16);
                b.f4[0] = bp[0];
                b.f4[1] = bp[1];
                __builtin_prefetch(bRow + 32, 0, 3);
                acc[j] = __builtin_amdgcn_wmma_f32_16x16x32_bf16(
                    false, a.v, false, b.v, (short)0, acc[j], false, false);
            }
        }
    }
#pragma unroll
    for (int j = 0; j < 4; ++j) {
        const int n = nBase + j * 16 + lh;
        const float bv = bias ? bias[n] : 0.f;
#pragma unroll
        for (int v = 0; v < 8; ++v) {
            size_t row = aRow0 + rw * 16 + v + 8 * half;
            float val = acc[j][v] + bv;
            if (relu) val = fmaxf(val, 0.f);
            if (outF) outF[row * Nn + n] = val;
            else      outH[row * Nn + n] = (bf16)val;
        }
    }
}

// ---------------------------------------------------------------------------
// Flash attention with WMMA (QK^T and P*V both on the matrix unit).
// grid = (seq, head, qblock); block = 256 (8 waves x 16 queries = 128 q/blk).
// Keys processed in 128-row chunks with online softmax. The K chunk is staged
// into LDS with GLOBAL_LOAD_ASYNC_TO_LDS_B128 (ASYNCcnt path, no VGPR
// round-trip); V is transposed through VGPRs; P converted D->A layout via LDS.
// All attention LDS lives in ONE __shared__ block so the async-load LDS byte
// offsets are well defined (sole shared allocation => base offset 0, K at 0).
// maskMode: 0 none, 1 stage1 (row = seq % 32, tile(tkpm) quirk), 2 cross.
// ---------------------------------------------------------------------------
#define SM_K_BYTES   (128 * DH * 2)                 // 8192
#define SM_VT_BYTES  (DH * 128 * 2)                 // 8192
#define SM_P_BYTES   (8 * 16 * 128 * 2)             // 32768
__global__ void flash_attn_wmma(const bf16* __restrict__ Q, const bf16* __restrict__ Kb,
                                const bf16* __restrict__ Vb, bf16* __restrict__ O,
                                const unsigned char* __restrict__ mask, int maskMode,
                                int nqPerSeq, int nkPerSeq, int nChunks) {
    __shared__ __align__(16) char smem[SM_K_BYTES + SM_VT_BYTES + SM_P_BYTES];
    bf16* smK  = (bf16*)smem;                                   // LDS byte off 0
    bf16* smVt = (bf16*)(smem + SM_K_BYTES);
    bf16* smP  = (bf16*)(smem + SM_K_BYTES + SM_VT_BYTES);

    const int seq = blockIdx.x, head = blockIdx.y, qb = blockIdx.z;
    const int tid = threadIdx.x;
    const int wave = tid >> 5, lane = tid & 31, lh = lane & 15, half = lane >> 4;
    const size_t qRow0 = (size_t)seq * nqPerSeq + qb * 128 + wave * 16;
    const size_t kvRow0 = (size_t)seq * nkPerSeq;

    Frag qf;
    {
        const bf16* qp = Q + (qRow0 + lh) * DD + head * DH;
        qf.f4[0] = *(const float4*)(qp + half * 16);
        qf.f4[1] = *(const float4*)(qp + half * 16 + 8);
    }
    const unsigned char* mrow = nullptr;
    if (maskMode == 1)      mrow = mask + (size_t)(seq % BSZ) * RR;
    else if (maskMode == 2) mrow = mask + (size_t)seq * NNM;

    float mi[8], li[8];
#pragma unroll
    for (int v = 0; v < 8; ++v) { mi[v] = -1e30f; li[v] = 0.f; }
    floatx8 accO0 = {0, 0, 0, 0, 0, 0, 0, 0};
    floatx8 accO1 = {0, 0, 0, 0, 0, 0, 0, 0};
    const float scale = 0.17677669529663687f; // 1/sqrt(32)
    bf16* pbase = smP + wave * 16 * 128;

    for (int ch = 0; ch < nChunks; ++ch) {
        const size_t kb0 = kvRow0 + ch * 128;
        __syncthreads();
        // K chunk: 128 rows x 64B, async direct-to-LDS (2 b128 per thread)
        for (int i = tid; i < 512; i += 256) {
            int row = i >> 2, seg = i & 3;
            unsigned long long src =
                (unsigned long long)(const void*)(Kb + (kb0 + row) * DD + head * DH + seg * 8);
            unsigned dst = (unsigned)(i * 16);          // byte offset of smK[..]
            asm volatile("global_load_async_to_lds_b128 %0, %1, off"
                         :: "v"(dst), "v"(src) : "memory");
        }
        // V transposed (needs in-flight transpose -> regular ds stores)
        for (int i = tid; i < 2048; i += 256) {
            int row = i >> 4, dp = i & 15;
            const bf16* vp = Vb + (kb0 + row) * DD + head * DH + dp * 2;
            smVt[(dp * 2) * 128 + row]     = vp[0];
            smVt[(dp * 2 + 1) * 128 + row] = vp[1];
        }
        asm volatile("s_wait_asynccnt 0" ::: "memory");
        __syncthreads();

        floatx8 S[8];
#pragma unroll
        for (int t = 0; t < 8; ++t) {
            Frag kf;
            const float4* kp = (const float4*)(smK + (t * 16 + lh) * DH + half * 16);
            kf.f4[0] = kp[0];
            kf.f4[1] = kp[1];
            floatx8 z = {0, 0, 0, 0, 0, 0, 0, 0};
            S[t] = __builtin_amdgcn_wmma_f32_16x16x32_bf16(false, qf.v, false, kf.v,
                                                           (short)0, z, false, false);
        }
#pragma unroll
        for (int t = 0; t < 8; ++t) {
            float pen = 0.f;
            if (mrow) pen = mrow[ch * 128 + t * 16 + lh] ? -1e9f : 0.f;
#pragma unroll
            for (int v = 0; v < 8; ++v) S[t][v] = S[t][v] * scale + pen;
        }
#pragma unroll
        for (int v = 0; v < 8; ++v) {
            float r = -1e30f;
#pragma unroll
            for (int t = 0; t < 8; ++t) r = fmaxf(r, S[t][v]);
#pragma unroll
            for (int m = 8; m >= 1; m >>= 1) r = fmaxf(r, __shfl_xor(r, m));
            float mn = fmaxf(mi[v], r);
            float alpha = __expf(mi[v] - mn);
            float rs = 0.f;
#pragma unroll
            for (int t = 0; t < 8; ++t) { float p = __expf(S[t][v] - mn); S[t][v] = p; rs += p; }
#pragma unroll
            for (int m = 8; m >= 1; m >>= 1) rs += __shfl_xor(rs, m);
            li[v] = li[v] * alpha + rs;
            mi[v] = mn;
            accO0[v] *= alpha;
            accO1[v] *= alpha;
        }
        // P: D-layout (row = v + 8*half, col = t*16 + lh) -> LDS row-major
#pragma unroll
        for (int t = 0; t < 8; ++t)
#pragma unroll
            for (int v = 0; v < 8; ++v)
                pbase[(v + 8 * half) * 128 + t * 16 + lh] = (bf16)S[t][v];
        __syncthreads();
#pragma unroll
        for (int kt = 0; kt < 4; ++kt) {
            Frag pf, vf0, vf1;
            const bf16* pp = pbase + lh * 128 + kt * 32 + half * 16;
            pf.f4[0] = *(const float4*)pp;
            pf.f4[1] = *(const float4*)(pp + 8);
            const float4* vp0 = (const float4*)(smVt + lh * 128 + kt * 32 + half * 16);
            vf0.f4[0] = vp0[0];
            vf0.f4[1] = vp0[1];
            const float4* vp1 = (const float4*)(smVt + (16 + lh) * 128 + kt * 32 + half * 16);
            vf1.f4[0] = vp1[0];
            vf1.f4[1] = vp1[1];
            accO0 = __builtin_amdgcn_wmma_f32_16x16x32_bf16(false, pf.v, false, vf0.v,
                                                            (short)0, accO0, false, false);
            accO1 = __builtin_amdgcn_wmma_f32_16x16x32_bf16(false, pf.v, false, vf1.v,
                                                            (short)0, accO1, false, false);
        }
    }
#pragma unroll
    for (int v = 0; v < 8; ++v) {
        float invl = 1.f / li[v];
        size_t row = qRow0 + v + 8 * half;
        bf16* orow = O + row * DD + head * DH;
        orow[lh]      = (bf16)(accO0[v] * invl);
        orow[16 + lh] = (bf16)(accO1[v] * invl);
    }
}

// ---------------------------------------------------------------------------
// m2m attention: seqlen 6, negligible FLOPs -> plain VALU kernel.
// thread = ((b*R + r)*H + h)*6 + q
// ---------------------------------------------------------------------------
__global__ void m2m_attn(const bf16* __restrict__ Q, const bf16* __restrict__ Kb,
                         const bf16* __restrict__ Vb, bf16* __restrict__ O) {
    int idx = blockIdx.x * 256 + threadIdx.x;
    if (idx >= BSZ * RR * HHD * MMD) return;
    int q = idx % MMD;
    int tmp = idx / MMD;
    int h = tmp % HHD;
    int br = tmp / HHD;
    size_t tb = (size_t)br * MMD;
    const bf16* qp = Q + (tb + q) * DD + h * DH;
    float qv[32];
#pragma unroll
    for (int d = 0; d < 32; ++d) qv[d] = (float)qp[d];
    float s[MMD], mx = -1e30f;
    for (int k = 0; k < MMD; ++k) {
        const bf16* kp = Kb + (tb + k) * DD + h * DH;
        float a = 0.f;
#pragma unroll
        for (int d = 0; d < 32; ++d) a += qv[d] * (float)kp[d];
        s[k] = a * 0.17677669529663687f;
        mx = fmaxf(mx, s[k]);
    }
    float se = 0.f;
    for (int k = 0; k < MMD; ++k) { s[k] = __expf(s[k] - mx); se += s[k]; }
    const float inv = 1.f / se;
    float o[32];
#pragma unroll
    for (int d = 0; d < 32; ++d) o[d] = 0.f;
    for (int k = 0; k < MMD; ++k) {
        const bf16* vp = Vb + (tb + k) * DD + h * DH;
        float a = s[k] * inv;
#pragma unroll
        for (int d = 0; d < 32; ++d) o[d] += a * (float)vp[d];
    }
    bf16* op = O + (tb + q) * DD + h * DH;
#pragma unroll
    for (int d = 0; d < 32; ++d) op[d] = (bf16)o[d];
}

// ---------------------------------------------------------------------------
// Epilogue / residual kernels (float4 vectorized).
// ---------------------------------------------------------------------------
__global__ void scatter_s1(float* __restrict__ X, const float* __restrict__ G) {
    int gid = blockIdx.x * 256 + threadIdx.x;          // over TT*64 float4s
    if (gid >= TT * 64) return;
    int s = gid >> 6, chunk = gid & 63;
    int b = s / (MMD * RR);
    int rem = s - b * (MMD * RR);
    int mi = rem / RR;
    int r = rem - mi * RR;
    int c = (b * RR + r) * MMD + mi;
    float4 gv = ((const float4*)G)[gid];
    float4* xp = (float4*)X + (size_t)c * 64 + chunk;
    float4 xv = *xp;
    xv.x += gv.x; xv.y += gv.y; xv.z += gv.z; xv.w += gv.w;
    *xp = xv;
}

__global__ void scatter_m2m(float* __restrict__ X, const float* __restrict__ G,
                            const unsigned char* __restrict__ tkpm) {
    int gid = blockIdx.x * 256 + threadIdx.x;
    if (gid >= TT * 64) return;
    int t = gid >> 6;
    int br = t / MMD;                    // b*R + r
    bool valid = tkpm[br] == 0;
    float4 o;
    if (valid) {
        float4 gv = ((const float4*)G)[gid];
        float4 xv = ((const float4*)X)[gid];
        o.x = xv.x + gv.x; o.y = xv.y + gv.y; o.z = xv.z + gv.z; o.w = xv.w + gv.w;
    } else {
        o.x = 0.f; o.y = 0.f; o.z = 0.f; o.w = 0.f;
    }
    ((float4*)X)[gid] = o;
}

__global__ void add_res(float* __restrict__ X, const float* __restrict__ G) {
    int gid = blockIdx.x * 256 + threadIdx.x;
    if (gid >= TT * 64) return;
    float4 gv = ((const float4*)G)[gid];
    float4 xv = ((const float4*)X)[gid];
    xv.x += gv.x; xv.y += gv.y; xv.z += gv.z; xv.w += gv.w;
    ((float4*)X)[gid] = xv;
}

__global__ void final_add(const float* __restrict__ X, const float* __restrict__ G,
                          float* __restrict__ out) {
    int gid = blockIdx.x * 256 + threadIdx.x;
    if (gid >= TT * 64) return;
    float4 gv = ((const float4*)G)[gid];
    float4 xv = ((const float4*)X)[gid];
    float4 o;
    o.x = xv.x + gv.x; o.y = xv.y + gv.y; o.z = xv.z + gv.z; o.w = xv.w + gv.w;
    ((float4*)out)[gid] = o;
}

// ---------------------------------------------------------------------------
extern "C" void kernel_launch(void* const* d_in, const int* in_sizes, int n_in,
                              void* d_out, int out_size, void* d_ws, size_t ws_size,
                              hipStream_t stream) {
    if (n_in < 41) return;
    const float* tgt    = (const float*)d_in[0];
    const float* memory = (const float*)d_in[1];
    const float* m_pos  = (const float*)d_in[2];
    const float *Wm[3][4], *Bm[3][4];
    const int base[3] = {3, 11, 19};     // r2r, m2m, cross
    for (int i = 0; i < 3; ++i)
        for (int j = 0; j < 4; ++j) {
            Wm[i][j] = (const float*)d_in[base[i] + j];
            Bm[i][j] = (const float*)d_in[base[i] + 4 + j];
        }
    const float* W1 = (const float*)d_in[27];
    const float* b1 = (const float*)d_in[28];
    const float* W2 = (const float*)d_in[29];
    const float* b2 = (const float*)d_in[30];
    const float *lng[4], *lnb[4];
    for (int i = 0; i < 4; ++i) {
        lng[i] = (const float*)d_in[31 + 2 * i];
        lnb[i] = (const float*)d_in[32 + 2 * i];
    }
    const unsigned char* tkpm = (const unsigned char*)d_in[39];
    const unsigned char* mkpm = (const unsigned char*)d_in[40];

    // ---- workspace carve (~137 MB) ----
    char* ws = (char*)d_ws;
    size_t off = 0;
    auto carve = [&](size_t bytes) -> void* {
        void* p = ws + off;
        off += (bytes + 255) & ~(size_t)255;
        return p;
    };
    float* X = (float*)carve((size_t)TT * DD * 4);
    bf16* WT[3][4];
    for (int i = 0; i < 3; ++i)
        for (int j = 0; j < 4; ++j) WT[i][j] = (bf16*)carve((size_t)DD * DD * 2);
    bf16* W1T  = (bf16*)carve((size_t)DD * MLPD * 2);
    bf16* W2T  = (bf16*)carve((size_t)MLPD * DD * 2);
    bf16* memb = (bf16*)carve((size_t)TMEM * DD * 2);
    bf16* A1   = (bf16*)carve((size_t)TT * DD * 2);
    bf16* A2p  = (bf16*)carve((size_t)TT * DD * 2);
    bf16* Qb   = (bf16*)carve((size_t)TT * DD * 2);
    bf16* Kbuf = (bf16*)carve((size_t)TT * DD * 2);
    bf16* Vbuf = (bf16*)carve((size_t)TT * DD * 2);
    bf16* Ob   = (bf16*)carve((size_t)TT * DD * 2);
    float* Gf  = (float*)carve((size_t)TT * DD * 4);
    bf16* Hb = Qb; // FFN hidden (TT x 1024 bf16) aliases Qb..Ob (attention dead then)

    const dim3 blk(256);
    const int eltGrid = (TT * 64 + 255) / 256;

    auto gemm = [&](const bf16* A, const bf16* Bt, const float* bias,
                    float* oF, bf16* oH, int Mtot, int Nn, int K, int relu) {
        dim3 g(Mtot / 32, Nn / 256);
        gemm_bf16_wmma<<<g, blk, 0, stream>>>(A, Bt, bias, oF, oH, K, Nn, relu);
    };

    // ---- weight/memory prep ----
    for (int i = 0; i < 3; ++i)
        for (int j = 0; j < 4; ++j)
            wconv_T<<<(DD * DD + 255) / 256, blk, 0, stream>>>(Wm[i][j], WT[i][j], DD, DD);
    wconv_T<<<(DD * MLPD + 255) / 256, blk, 0, stream>>>(W1, W1T, DD, MLPD);
    wconv_T<<<(MLPD * DD + 255) / 256, blk, 0, stream>>>(W2, W2T, MLPD, DD);
    f2bf<<<(TMEM * DD + 255) / 256, blk, 0, stream>>>(memory, memb, TMEM * DD);
    hipMemcpyAsync(X, tgt, (size_t)TT * DD * 4, hipMemcpyDeviceToDevice, stream);

    // ---- stage 1: r2r self-attention (tokens in (b,m,r) order) ----
    ln_fused<<<TT / 8, blk, 0, stream>>>(X, lng[0], lnb[0], A1, nullptr, nullptr, 1);
    gemm(A1, WT[0][0], Bm[0][0], nullptr, Qb,   TT, DD, DD, 0);
    gemm(A1, WT[0][1], Bm[0][1], nullptr, Kbuf, TT, DD, DD, 0);
    gemm(A1, WT[0][2], Bm[0][2], nullptr, Vbuf, TT, DD, DD, 0);
    flash_attn_wmma<<<dim3(BSZ * MMD, HHD, 1), blk, 0, stream>>>(
        Qb, Kbuf, Vbuf, Ob, tkpm, 1, RR, RR, 1);
    gemm(Ob, WT[0][3], Bm[0][3], Gf, nullptr, TT, DD, DD, 0);
    scatter_s1<<<eltGrid, blk, 0, stream>>>(X, Gf);

    // ---- stage 2: m2m attention (canonical token order) ----
    ln_fused<<<TT / 8, blk, 0, stream>>>(X, lng[1], lnb[1], A1, A2p, m_pos, 0);
    gemm(A2p, WT[1][0], Bm[1][0], nullptr, Qb,   TT, DD, DD, 0);
    gemm(A2p, WT[1][1], Bm[1][1], nullptr, Kbuf, TT, DD, DD, 0);
    gemm(A1,  WT[1][2], Bm[1][2], nullptr, Vbuf, TT, DD, DD, 0);
    m2m_attn<<<(BSZ * RR * HHD * MMD + 255) / 256, blk, 0, stream>>>(Qb, Kbuf, Vbuf, Ob);
    gemm(Ob, WT[1][3], Bm[1][3], Gf, nullptr, TT, DD, DD, 0);
    scatter_m2m<<<eltGrid, blk, 0, stream>>>(X, Gf, tkpm);

    // ---- stage 3: cross attention to memory ----
    ln_fused<<<TT / 8, blk, 0, stream>>>(X, lng[2], lnb[2], A1, nullptr, nullptr, 0);
    gemm(A1,   WT[2][0], Bm[2][0], nullptr, Qb,   TT,   DD, DD, 0);
    gemm(memb, WT[2][1], Bm[2][1], nullptr, Kbuf, TMEM, DD, DD, 0);
    gemm(memb, WT[2][2], Bm[2][2], nullptr, Vbuf, TMEM, DD, DD, 0);
    flash_attn_wmma<<<dim3(BSZ, HHD, 6), blk, 0, stream>>>(
        Qb, Kbuf, Vbuf, Ob, mkpm, 2, RR * MMD, NNM, 4);
    gemm(Ob, WT[2][3], Bm[2][3], Gf, nullptr, TT, DD, DD, 0);
    add_res<<<eltGrid, blk, 0, stream>>>(X, Gf);

    // ---- FFN ----
    ln_fused<<<TT / 8, blk, 0, stream>>>(X, lng[3], lnb[3], A1, nullptr, nullptr, 0);
    gemm(A1, W1T, b1, nullptr, Hb, TT, MLPD, DD, 1);
    gemm(Hb, W2T, b2, Gf, nullptr, TT, DD, MLPD, 0);
    final_add<<<eltGrid, blk, 0, stream>>>(X, Gf, (float*)d_out);
}
